// SummaryNetwork_36962488550104
// MI455X (gfx1250) — matrix operand order repaired and checked by reading, not compiled, not run on hardware
//
#include <hip/hip_runtime.h>
#include <hip/hip_bf16.h>

// ---------------------------------------------------------------------------
// 2-layer LSTM, B=64, T=512, I=128, H=512, gate order i,f,g,o.
//   Phase A: xg = input @ W_ih^T + (b_ih + b_hh)   (big WMMA GEMM, f16->f32)
//   Phase B: persistent recurrent kernel, 32 WGs, device-wide barrier per step
//            gates = xg[:,t,:] + h @ W_hh^T  (WMMA), then elementwise c/h.
//   W_hh fragments live in VGPRs across all 512 steps; xg is double-buffered
//   into LDS with global_load_async_to_lds_b128 + s_wait_asynccnt when the
//   toolchain exposes the gfx1250 async builtins.
// ---------------------------------------------------------------------------

typedef __attribute__((ext_vector_type(16))) _Float16 v16h;
typedef __attribute__((ext_vector_type(8)))  _Float16 v8h;
typedef __attribute__((ext_vector_type(8)))  float    v8f;
typedef int a_i4 __attribute__((vector_size(16)));   // matches builtin param type

#define I_SIZE 128
#define HDIM   512
#define G4     2048              // 4*H
#define BATCH  64
#define TSTEPS 512
#define MROWS  (BATCH * TSTEPS)  // 32768
#define NWG    32                // persistent workgroups (each owns 16 h-cols)

#if defined(__has_builtin)
#if __has_builtin(__builtin_amdgcn_global_load_async_to_lds_b128) && \
    __has_builtin(__builtin_amdgcn_s_wait_asynccnt)
#define USE_ASYNC 1
#endif
#endif
#ifndef USE_ASYNC
#define USE_ASYNC 0
#endif

// ---- workspace layout (bytes) ---------------------------------------------
static constexpr size_t OFF_XG   = 0;                                     // [MROWS, G4] f32
static constexpr size_t OFF_XH   = OFF_XG   + (size_t)MROWS * G4 * 4;     // x as f16 [MROWS, I]
static constexpr size_t OFF_H0O  = OFF_XH   + (size_t)MROWS * I_SIZE * 2; // layer0 h seq f16 [MROWS, H]
static constexpr size_t OFF_WIH0 = OFF_H0O  + (size_t)MROWS * HDIM * 2;   // [G4, I] f16
static constexpr size_t OFF_WHH0 = OFF_WIH0 + (size_t)G4 * I_SIZE * 2;    // [G4, H] f16
static constexpr size_t OFF_WIH1 = OFF_WHH0 + (size_t)G4 * HDIM * 2;      // [G4, H] f16
static constexpr size_t OFF_WHH1 = OFF_WIH1 + (size_t)G4 * HDIM * 2;      // [G4, H] f16
static constexpr size_t OFF_B0   = OFF_WHH1 + (size_t)G4 * HDIM * 2;      // [G4] f32
static constexpr size_t OFF_B1   = OFF_B0   + (size_t)G4 * 4;             // [G4] f32
static constexpr size_t OFF_HB0  = OFF_B1   + (size_t)G4 * 4;             // layer0 h double-buffer f16 [2][B*H]
static constexpr size_t OFF_HB1  = OFF_HB0  + (size_t)2 * BATCH * HDIM * 2;
static constexpr size_t OFF_CNT  = OFF_HB1  + (size_t)2 * BATCH * HDIM * 2; // 2 barrier counters

// ---- helpers --------------------------------------------------------------
static __device__ __forceinline__ v8h ld8(const _Float16* p) {
  return *reinterpret_cast<const v8h*>(p);
}
static __device__ __forceinline__ v16h mk16(v8h lo, v8h hi) {
  v16h r;
#pragma unroll
  for (int i = 0; i < 8; ++i) { r[i] = lo[i]; r[i + 8] = hi[i]; }
  return r;
}
static __device__ __forceinline__ v8f wmma_f16(v16h a, v16h b, v8f c) {
  // D = A(16x32 f16) * B(32x16 f16) + C(16x16 f32)
  return __builtin_amdgcn_wmma_f32_16x16x32_f16(false, a, false, b, (short)0, c,
                                                false, false);
}
static __device__ __forceinline__ float sigm(float x) {
  return 1.0f / (1.0f + __expf(-x));
}

// ---- small utility kernels -------------------------------------------------
__global__ void k_cvt_f64_f16(const double* __restrict__ s, _Float16* __restrict__ d, int n) {
  for (int i = blockIdx.x * blockDim.x + threadIdx.x; i < n; i += gridDim.x * blockDim.x)
    d[i] = (_Float16)s[i];
}
__global__ void k_cvt_f32_f16(const float* __restrict__ s, _Float16* __restrict__ d, int n) {
  for (int i = blockIdx.x * blockDim.x + threadIdx.x; i < n; i += gridDim.x * blockDim.x)
    d[i] = (_Float16)s[i];
}
__global__ void k_bias_sum(const double* __restrict__ a, const double* __restrict__ b,
                           float* __restrict__ d, int n) {
  for (int i = blockIdx.x * blockDim.x + threadIdx.x; i < n; i += gridDim.x * blockDim.x)
    d[i] = (float)(a[i] + b[i]);
}
__global__ void k_zero_f16(_Float16* __restrict__ d, int n) {
  for (int i = blockIdx.x * blockDim.x + threadIdx.x; i < n; i += gridDim.x * blockDim.x)
    d[i] = (_Float16)0.0f;
}
__global__ void k_zero_u32(unsigned* __restrict__ d, int n) {
  for (int i = blockIdx.x * blockDim.x + threadIdx.x; i < n; i += gridDim.x * blockDim.x)
    d[i] = 0u;
}
__global__ void k_zero_out_col0(float* __restrict__ out) {
  int i = blockIdx.x * blockDim.x + threadIdx.x;   // 64*512 elements
  if (i < BATCH * HDIM) {
    int b = i >> 9, c = i & 511;
    out[(long long)b * (TSTEPS + 1) * HDIM + c] = 0.0f;
  }
}

// ---- Phase A: xg = A[M,K](f16) @ W[G4,K]^T(f16) + bias, C f32 [M,G4] -------
__global__ __launch_bounds__(256) void k_gemm_xg(const _Float16* __restrict__ A,
                                                 const _Float16* __restrict__ W,
                                                 const float* __restrict__ bias,
                                                 float* __restrict__ C, int K) {
  const int lane = threadIdx.x & 31;
  const int wv   = threadIdx.x >> 5;
  const int l16  = lane & 15;
  const bool hi  = lane >= 16;

  const long long tile = (long long)blockIdx.x * 8 + wv;  // 262144 tiles total
  const int MT = MROWS / 16;                              // 2048 (pow2)
  const int mt = (int)(tile & (MT - 1));
  const int nt = (int)(tile >> 11);                       // tile / 2048, nt < 128

  const _Float16* Ap = A + (long long)(mt * 16 + l16) * K;
  const _Float16* Bp = W + (long long)(nt * 16 + l16) * K;
  const int ka = hi ? 8 : 0;    // A frag K sub-offset per ISA layout
  const int kb = hi ? 8 : 0;    // B frag K sub-offset (same order as A)

  v8f acc = {};
  for (int k0 = 0; k0 < K; k0 += 32) {
    v16h a = mk16(ld8(Ap + k0 + ka), ld8(Ap + k0 + ka + 16));
    v16h b = mk16(ld8(Bp + k0 + kb), ld8(Bp + k0 + kb + 16));
    acc = wmma_f16(a, b, acc);
  }

  const int nc = nt * 16 + l16;
  const float bv = bias[nc];
  const int mb = mt * 16 + (hi ? 8 : 0);
#pragma unroll
  for (int r = 0; r < 8; ++r)
    C[(long long)(mb + r) * G4 + nc] = acc[r] + bv;
}

// ---- Phase B: persistent recurrent layer -----------------------------------
// grid = NWG blocks x 256 threads (8 wave32). WG `wg` owns h-cols
// [wg*16, wg*16+16) and the matching 4 gate column groups {g*512 + wg*16}.
// Wave w: gate gi = w&3, M-tile pair = w>>2 (M=64 -> 4 M-tiles).
__global__ __launch_bounds__(256) void k_lstm_layer(
    const float* __restrict__ xg,       // [MROWS, G4], row = b*T + t
    const _Float16* __restrict__ Whh,   // [G4, H] row-major
    _Float16* __restrict__ hbuf,        // [2][B*H] double buffer (buf0 zeroed)
    _Float16* __restrict__ hseq,        // layer0: [MROWS, H] f16 output (or null)
    float* __restrict__ outp,           // layer1: [B, T+1, H] f32 output (or null)
    unsigned* __restrict__ bar) {
  const int wg   = blockIdx.x;
  const int tid  = threadIdx.x;
  const int lane = tid & 31;
  const int wv   = tid >> 5;
  const int l16  = lane & 15;
  const bool hi  = lane >= 16;
  const int gi    = wv & 3;   // gate index i/f/g/o
  const int mpair = wv >> 2;  // handles M-tiles {2*mpair, 2*mpair+1}

  __shared__ float sG[64][68];  // staged gates: cols [gi*16 .. gi*16+15]
  __shared__ float sC[64][16];  // persistent cell state for this WG's slice
#if USE_ASYNC
  __shared__ float sXG[2][64][64];  // double-buffered xg tile (async-filled)
#endif

  for (int u = tid; u < 64 * 16; u += 256) sC[u >> 4][u & 15] = 0.0f;

  // Hoist W_hh B-fragments into VGPRs: invariant across all 512 steps.
  const int n = gi * HDIM + wg * 16 + l16;
  const _Float16* Bp = Whh + (long long)n * HDIM;
  const int kb = hi ? 8 : 0;
  const int ka = hi ? 8 : 0;
  v16h bf[16];
#pragma unroll
  for (int kk = 0; kk < 16; ++kk)
    bf[kk] = mk16(ld8(Bp + kk * 32 + kb), ld8(Bp + kk * 32 + kb + 16));

  const int m0 = (mpair * 2) * 16 + l16;   // A frag row, M-tile 0
  const int m1 = m0 + 16;                  // A frag row, M-tile 1

#if USE_ASYNC
  // Preload xg tile for t=0 into buffer 0. Each thread moves 4x16B.
#pragma unroll
  for (int u = 0; u < 4; ++u) {
    const int li = tid + 256 * u;
    const int f = li * 4;
    const int b = f >> 6, c = f & 63;
    const int gate = c >> 4, c0 = c & 15;
    const float* g = xg + ((long long)b * TSTEPS + 0) * G4 + gate * HDIM + wg * 16 + c0;
    __builtin_amdgcn_global_load_async_to_lds_b128(
        (a_i4*)(void*)g, (a_i4*)(void*)&sXG[0][b][c], 0, 0);
  }
#endif
  __syncthreads();

  for (int t = 0; t < TSTEPS; ++t) {
#if USE_ASYNC
    // Issue async fill of next step's xg tile; overlaps with the WMMA chain.
    {
      const int tn = (t + 1 < TSTEPS) ? (t + 1) : t;
#pragma unroll
      for (int u = 0; u < 4; ++u) {
        const int li = tid + 256 * u;
        const int f = li * 4;
        const int b = f >> 6, c = f & 63;
        const int gate = c >> 4, c0 = c & 15;
        const float* g = xg + ((long long)b * TSTEPS + tn) * G4 + gate * HDIM + wg * 16 + c0;
        __builtin_amdgcn_global_load_async_to_lds_b128(
            (a_i4*)(void*)g, (a_i4*)(void*)&sXG[(t + 1) & 1][b][c], 0, 0);
      }
    }
#endif

    const _Float16* hin = hbuf + (size_t)(t & 1) * (BATCH * HDIM);
    v8f acc0 = {}, acc1 = {};
#pragma unroll
    for (int kk = 0; kk < 16; ++kk) {
      const int k0 = kk * 32;
      v16h a0 = mk16(ld8(hin + m0 * HDIM + k0 + ka),
                     ld8(hin + m0 * HDIM + k0 + ka + 16));
      v16h a1 = mk16(ld8(hin + m1 * HDIM + k0 + ka),
                     ld8(hin + m1 * HDIM + k0 + ka + 16));
      acc0 = wmma_f16(a0, bf[kk], acc0);
      acc1 = wmma_f16(a1, bf[kk], acc1);
    }

    // stage gate partials to LDS (C-frag layout: VGPR r -> M=r / r+8)
    const int row0 = mpair * 32 + (hi ? 8 : 0);
    const int col  = gi * 16 + l16;
#pragma unroll
    for (int r = 0; r < 8; ++r) {
      sG[row0 + r][col]      = acc0[r];
      sG[row0 + 16 + r][col] = acc1[r];
    }
#if USE_ASYNC
    // Oldest 4 async groups (step t's tile) complete; newest 4 stay in flight.
    __builtin_amdgcn_s_wait_asynccnt(4);
#endif
    __syncthreads();

    // elementwise: 64 batches x 16 h-cols, 4 per thread
    _Float16* hnext = hbuf + (size_t)((t + 1) & 1) * (BATCH * HDIM);
#pragma unroll
    for (int u = 0; u < 4; ++u) {
      const int e = tid + 256 * u;
      const int b = e >> 4, c = e & 15;
#if USE_ASYNC
      const float* xr = &sXG[t & 1][b][0];
      float iv = sG[b][c]      + xr[c];
      float fv = sG[b][16 + c] + xr[16 + c];
      float gv = sG[b][32 + c] + xr[32 + c];
      float ov = sG[b][48 + c] + xr[48 + c];
#else
      const long long xrow = ((long long)b * TSTEPS + t) * G4 + wg * 16 + c;
      float iv = sG[b][c]      + xg[xrow];
      float fv = sG[b][16 + c] + xg[xrow + 512];
      float gv = sG[b][32 + c] + xg[xrow + 1024];
      float ov = sG[b][48 + c] + xg[xrow + 1536];
#endif
      float cc = sigm(fv) * sC[b][c] + sigm(iv) * tanhf(gv);
      float hh = sigm(ov) * tanhf(cc);
      sC[b][c] = cc;
      const int hcol = wg * 16 + c;
      hnext[b * HDIM + hcol] = (_Float16)hh;
      if (hseq) hseq[((long long)b * TSTEPS + t) * HDIM + hcol] = (_Float16)hh;
      if (outp) outp[((long long)b * (TSTEPS + 1) + (t + 1)) * HDIM + hcol] = hh;
    }

#if !USE_ASYNC
    // warm next step's xg rows into cache (global_prefetch_b8)
    if (t + 1 < TSTEPS && tid < BATCH)
      __builtin_prefetch(xg + ((long long)tid * TSTEPS + t + 1) * G4 + wg * 16, 0, 1);
#endif

    // device-wide barrier (counter zeroed per launch; release h stores)
    __syncthreads();
    if (tid == 0) {
      __hip_atomic_fetch_add(bar, 1u, __ATOMIC_RELEASE, __HIP_MEMORY_SCOPE_AGENT);
      const unsigned target = (unsigned)(t + 1) * NWG;
      while (__hip_atomic_load(bar, __ATOMIC_ACQUIRE, __HIP_MEMORY_SCOPE_AGENT) < target)
        __builtin_amdgcn_s_sleep(2);
    }
    __syncthreads();
  }
}

// ---------------------------------------------------------------------------
extern "C" void kernel_launch(void* const* d_in, const int* in_sizes, int n_in,
                              void* d_out, int out_size, void* d_ws, size_t ws_size,
                              hipStream_t stream) {
  (void)in_sizes; (void)n_in; (void)out_size; (void)ws_size;

  const float*  x    = (const float*) d_in[0];
  const double* Wih0 = (const double*)d_in[1];
  const double* Whh0 = (const double*)d_in[2];
  const double* bih0 = (const double*)d_in[3];
  const double* bhh0 = (const double*)d_in[4];
  const double* Wih1 = (const double*)d_in[5];
  const double* Whh1 = (const double*)d_in[6];
  const double* bih1 = (const double*)d_in[7];
  const double* bhh1 = (const double*)d_in[8];
  float* out = (float*)d_out;

  char* ws = (char*)d_ws;
  float*    xg    = (float*)   (ws + OFF_XG);
  _Float16* xh    = (_Float16*)(ws + OFF_XH);
  _Float16* h0out = (_Float16*)(ws + OFF_H0O);
  _Float16* wih0h = (_Float16*)(ws + OFF_WIH0);
  _Float16* whh0h = (_Float16*)(ws + OFF_WHH0);
  _Float16* wih1h = (_Float16*)(ws + OFF_WIH1);
  _Float16* whh1h = (_Float16*)(ws + OFF_WHH1);
  float*    bias0 = (float*)   (ws + OFF_B0);
  float*    bias1 = (float*)   (ws + OFF_B1);
  _Float16* hb0   = (_Float16*)(ws + OFF_HB0);
  _Float16* hb1   = (_Float16*)(ws + OFF_HB1);
  unsigned* cnt   = (unsigned*)(ws + OFF_CNT);

  // ---- init / precision conversion ----
  k_cvt_f32_f16<<<4096, 256, 0, stream>>>(x, xh, MROWS * I_SIZE);
  k_cvt_f64_f16<<<1024, 256, 0, stream>>>(Wih0, wih0h, G4 * I_SIZE);
  k_cvt_f64_f16<<<4096, 256, 0, stream>>>(Whh0, whh0h, G4 * HDIM);
  k_cvt_f64_f16<<<4096, 256, 0, stream>>>(Wih1, wih1h, G4 * HDIM);
  k_cvt_f64_f16<<<4096, 256, 0, stream>>>(Whh1, whh1h, G4 * HDIM);
  k_bias_sum<<<8, 256, 0, stream>>>(bih0, bhh0, bias0, G4);
  k_bias_sum<<<8, 256, 0, stream>>>(bih1, bhh1, bias1, G4);
  k_zero_f16<<<256, 256, 0, stream>>>(hb0, 2 * BATCH * HDIM);   // h0 = 0
  k_zero_f16<<<256, 256, 0, stream>>>(hb1, 2 * BATCH * HDIM);
  k_zero_u32<<<1, 64, 0, stream>>>(cnt, 2);
  k_zero_out_col0<<<(BATCH * HDIM + 255) / 256, 256, 0, stream>>>(out);

  const int gemm_blocks = (MROWS / 16) * (G4 / 16) / 8;  // 32768

  // ---- layer 0 ----
  k_gemm_xg<<<gemm_blocks, 256, 0, stream>>>(xh, wih0h, bias0, xg, I_SIZE);
  k_lstm_layer<<<NWG, 256, 0, stream>>>(xg, whh0h, hb0, h0out, nullptr, cnt + 0);

  // ---- layer 1 ----
  k_gemm_xg<<<gemm_blocks, 256, 0, stream>>>(h0out, wih1h, bias1, xg, HDIM);
  k_lstm_layer<<<NWG, 256, 0, stream>>>(xg, whh1h, hb1, nullptr, out, cnt + 1);
}
